// COPContrastive_86492051407526
// MI455X (gfx1250) — matrix operand order repaired and checked by reading, not compiled, not run on hardware
//
#include <hip/hip_runtime.h>
#include <hip/hip_bf16.h>
#include <math.h>

typedef __attribute__((ext_vector_type(16))) _Float16 v16h;
typedef __attribute__((ext_vector_type(8)))  _Float16 v8h;
typedef __attribute__((ext_vector_type(8)))  float    v8f;
typedef __attribute__((ext_vector_type(4)))  unsigned int v4u;
typedef __attribute__((ext_vector_type(8)))  int v8i;
typedef __attribute__((ext_vector_type(4)))  int v4i;

#define BN  512
#define SN  128
#define S1  129
#define DM  128
#define NH  4
#define DH  32
#define DFF 512
#define KP  160              // keys padded to 160 (5 k-steps of 32)
#define LPP 128              // pairs padded 127 -> 128
#define KSIMS ((size_t)LPP * DM)   // 16384
#define TAUF 0.07f

#if __has_builtin(__builtin_amdgcn_tensor_load_to_lds)
#define HAVE_TDM 1
#else
#define HAVE_TDM 0
#endif

// ---------------- workspace layout (bytes, 256B aligned) ----------------
static const size_t SZ_TOK = (size_t)BN * S1 * DM * 4;     // f32 residual stream
static const size_t SZ_H   = (size_t)BN * S1 * DM * 2;     // f16 LN output
static const size_t SZ_QKV = (size_t)BN * S1 * 384 * 2;    // f16
static const size_t SZ_CTX = (size_t)BN * S1 * DM * 2;     // f16 attn context
static const size_t OFF_TOK = 0;
static const size_t OFF_H   = OFF_TOK + SZ_TOK;
static const size_t OFF_QKV = OFF_H + SZ_H;
static const size_t OFF_CTX = OFF_QKV + SZ_QKV;
static const size_t OFF_WQ  = OFF_CTX + SZ_CTX;            // WqkvT f16 [384][128]
static const size_t OFF_WO  = OFF_WQ + 384 * 128 * 2;      // WoT   f16 [128][128]
static const size_t OFF_W1T = OFF_WO + 128 * 128 * 2;      // W1T   f16 [512][128]
static const size_t OFF_W2T = OFF_W1T + 512 * 128 * 2;     // W2T   f16 [128][512]
static const size_t OFF_AF  = OFF_W2T + 128 * 512 * 2;     // Af f16 [512][16384]
static const size_t OFF_BF  = OFF_AF + (size_t)BN * KSIMS * 2;
static const size_t OFF_DEN = OFF_BF + (size_t)BN * KSIMS * 2;
static const size_t OFF_RL  = OFF_DEN + 2048;

// ---------------- helpers ----------------
__device__ __forceinline__ int imin(int a, int b) { return a < b ? a : b; }

__device__ __forceinline__ float wred_sum(float v) {
#pragma unroll
  for (int o = 16; o; o >>= 1) v += __shfl_xor(v, o, 32);
  return v;
}
__device__ __forceinline__ float wred_max(float v) {
#pragma unroll
  for (int o = 16; o; o >>= 1) v = fmaxf(v, __shfl_xor(v, o, 32));
  return v;
}

// A fragment (16x32, MxK): lane r=lane%16 is row M; lanes 0-15 hold K 0-7 & 16-23,
// lanes 16-31 hold K 8-15 & 24-31 (ISA 7.12.2 16-bit A layout).
__device__ __forceinline__ v16h frag_ldA(const _Float16* base, int ld) {
  const int lane = threadIdx.x & 31;
  const _Float16* p = base + (size_t)(lane & 15) * ld + ((lane >> 4) << 3);
  union { v16h v; v8h h[2]; } u;
  u.h[0] = *(const v8h*)(p);
  u.h[1] = *(const v8h*)(p + 16);
  return u.v;
}
__device__ __forceinline__ v16h frag_ldA_guard(const _Float16* base, int ld, int rowsValid) {
  const int lane = threadIdx.x & 31;
  const int r = lane & 15;
  union { v16h v; v8h h[2]; } u;
  if (r < rowsValid) {
    const _Float16* p = base + (size_t)r * ld + ((lane >> 4) << 3);
    u.h[0] = *(const v8h*)(p);
    u.h[1] = *(const v8h*)(p + 16);
  } else {
#pragma unroll
    for (int t = 0; t < 16; ++t) u.v[t] = (_Float16)0.0f;
  }
  return u.v;
}
// B fragment (32x16, KxN) from K-major (Bt = [N][K]) storage: lane n=lane%16 is col N;
// lanes 0-15 hold K 0-7 & 16-23, lanes 16-31 hold K 8-15 & 24-31 (same as A).
__device__ __forceinline__ v16h frag_ldB(const _Float16* baseT, int ld) {
  const int lane = threadIdx.x & 31;
  const _Float16* p = baseT + (size_t)(lane & 15) * ld + ((lane >> 4) << 3);
  union { v16h v; v8h h[2]; } u;
  u.h[0] = *(const v8h*)(p);
  u.h[1] = *(const v8h*)(p + 16);
  return u.v;
}
__device__ __forceinline__ v8f wmma16(v16h a, v16h b, v8f c) {
  return __builtin_amdgcn_wmma_f32_16x16x32_f16(false, a, false, b, (short)0, c, false, false);
}

// gfx1250 async global->LDS copy (16B per lane), tracked by ASYNCcnt.
// LDS byte offset = low 32 bits of the generic shared-pointer (aperture in high bits).
__device__ __forceinline__ void async_ld_b128(void* lds_ptr, const void* gptr) {
  unsigned lds_off = (unsigned)(uintptr_t)lds_ptr;
  asm volatile("global_load_async_to_lds_b128 %0, %1, off"
               :: "v"(lds_off), "v"((unsigned long long)(uintptr_t)gptr)
               : "memory");
}
__device__ __forceinline__ void wait_async0() {
  asm volatile("s_wait_asynccnt 0" ::: "memory");
}
__device__ __forceinline__ void wait_tensor0() {
#if __has_builtin(__builtin_amdgcn_s_wait_tensorcnt)
  __builtin_amdgcn_s_wait_tensorcnt((short)0);
#else
  asm volatile("s_wait_tensorcnt 0" ::: "memory");
#endif
}

// ---------------- kernels ----------------

// transpose weights to f16 K-major (Bt)
__global__ void k_prep(const float* __restrict__ Wqkv, const float* __restrict__ Wo,
                       const float* __restrict__ W1, const float* __restrict__ W2,
                       _Float16* WqkvT, _Float16* WoT, _Float16* W1T, _Float16* W2T) {
  int id = blockIdx.x * 256 + threadIdx.x;
  if (id < 49152) {            // WqkvT [384][128]
    int n = id / 128, k = id % 128;
    WqkvT[id] = (_Float16)Wqkv[k * 384 + n];
  } else if (id < 65536) {     // WoT [128][128]
    int i2 = id - 49152; int n = i2 / 128, k = i2 % 128;
    WoT[i2] = (_Float16)Wo[k * 128 + n];
  } else if (id < 131072) {    // W1T [512][128]
    int i3 = id - 65536; int n = i3 / 128, k = i3 % 128;
    W1T[i3] = (_Float16)W1[k * 512 + n];
  } else if (id < 196608) {    // W2T [128][512]
    int i4 = id - 131072; int n = i4 / 512, k = i4 % 512;
    W2T[i4] = (_Float16)W2[k * 128 + n];
  }
}

// tok[b][s][d] = (s==0? cls : emb[item]) + pos
__global__ void k_embed(const int* __restrict__ items, const float* __restrict__ emb,
                        const float* __restrict__ clst, const float* __restrict__ pos,
                        float* __restrict__ tok) {
  int b = blockIdx.x;
  for (int idx = threadIdx.x; idx < S1 * DM; idx += 256) {
    int s = idx >> 7, d = idx & 127;
    float v = (s == 0) ? clst[d] : emb[(size_t)items[b * SN + (s - 1)] * DM + d];
    tok[((size_t)b * S1 + s) * DM + d] = v + pos[s * DM + d];
  }
}

// h = LN(tok) -> f16 (one wave per token)
__global__ void k_ln1(const float* __restrict__ tok, _Float16* __restrict__ hg) {
  int b = blockIdx.y;
  int w = threadIdx.x >> 5, lane = threadIdx.x & 31;
  int s = blockIdx.x * 8 + w;
  if (s >= S1) return;
  const float* x = tok + ((size_t)b * S1 + s) * DM;
  float v4[4]; float sm = 0.f;
#pragma unroll
  for (int t = 0; t < 4; ++t) { v4[t] = x[lane + 32 * t]; sm += v4[t]; }
  float mean = wred_sum(sm) * (1.0f / 128.0f);
  float vs = 0.f;
#pragma unroll
  for (int t = 0; t < 4; ++t) { float d = v4[t] - mean; vs += d * d; }
  float inv = rsqrtf(wred_sum(vs) * (1.0f / 128.0f) + 1e-5f);
  _Float16* o = hg + ((size_t)b * S1 + s) * DM;
#pragma unroll
  for (int t = 0; t < 4; ++t) o[lane + 32 * t] = (_Float16)((v4[t] - mean) * inv);
}

// qkv = h @ Wqkv  (M=16 tile per block, N=384, K=128); 3 N-tiles per wave, 3 acc chains
__global__ void k_qkv(const _Float16* __restrict__ hg, const _Float16* __restrict__ WqkvT,
                      _Float16* __restrict__ qkv) {
  int b = blockIdx.y, mt0 = blockIdx.x * 16;
  int rowsValid = imin(S1 - mt0, 16);
  int w = threadIdx.x >> 5, lane = threadIdx.x & 31;
  const _Float16* abase = hg + ((size_t)b * S1 + mt0) * DM;
  v16h afr[4];
#pragma unroll
  for (int ks = 0; ks < 4; ++ks) afr[ks] = frag_ldA_guard(abase + ks * 32, DM, rowsValid);
  int n = lane & 15, mo = (lane >> 4) << 3;
  v8f cc[3] = {};
#pragma unroll
  for (int ks = 0; ks < 4; ++ks) {
    v16h b0 = frag_ldB(WqkvT + (size_t)((w)      * 16) * DM + ks * 32, DM);
    v16h b1 = frag_ldB(WqkvT + (size_t)((w + 8)  * 16) * DM + ks * 32, DM);
    v16h b2 = frag_ldB(WqkvT + (size_t)((w + 16) * 16) * DM + ks * 32, DM);
    cc[0] = wmma16(afr[ks], b0, cc[0]);
    cc[1] = wmma16(afr[ks], b1, cc[1]);
    cc[2] = wmma16(afr[ks], b2, cc[2]);
  }
#pragma unroll
  for (int q = 0; q < 3; ++q) {
    int nt = w + q * 8;
    _Float16* ob = qkv + ((size_t)b * S1 + mt0) * 384 + nt * 16 + n;
    if (rowsValid == 16) {
#pragma unroll
      for (int r = 0; r < 8; ++r) ob[(size_t)(mo + r) * 384] = (_Float16)cc[q][r];
    } else {
#pragma unroll
      for (int r = 0; r < 8; ++r) { int m = mo + r; if (m < rowsValid) ob[(size_t)m * 384] = (_Float16)cc[q][r]; }
    }
  }
}

// attention: one block per (mtile, head, batch). keys padded to KP=160.
// K-pane staged by the Tensor Data Mover (rows >= S1 zero-filled via tensor_dim OOB).
__global__ void k_attn(const _Float16* __restrict__ qkv, const int* __restrict__ items,
                       _Float16* __restrict__ ctx) {
  __shared__ __align__(16) _Float16 kbuf[KP * DH];   // [key][dh]  = Bt for scores
  __shared__ __align__(16) _Float16 vT[DH * KP];     // [dh][key]  = Bt for context
  __shared__ __align__(16) float    sbuf[16 * KP];
  __shared__ __align__(16) _Float16 pbuf[16 * KP];
  int mt0 = blockIdx.x * 16, hh = blockIdx.y, b = blockIdx.z;
  int rowsValid = imin(S1 - mt0, 16);
  int w = threadIdx.x >> 5, lane = threadIdx.x & 31;
  for (int t = threadIdx.x; t < KP * DH; t += 256) {
    int j = t >> 5, d = t & 31;
    _Float16 vv = (_Float16)0.f;
    if (j < S1) vv = qkv[((size_t)b * S1 + j) * 384 + 256 + hh * DH + d];
    vT[d * KP + j] = vv;
#if !HAVE_TDM
    _Float16 kv = (_Float16)0.f;
    if (j < S1) kv = qkv[((size_t)b * S1 + j) * 384 + 128 + hh * DH + d];
    kbuf[j * DH + d] = kv;
#endif
  }
#if HAVE_TDM
  if (w == 0) {
    // D#: 2D tile 32 x 160 of 2B elems; tensor dims 32 x 129 (rows >=129 -> OOB zeros);
    // row stride 384 elems; dest kbuf packed [160][32].
    unsigned long long ga =
        (unsigned long long)(uintptr_t)(qkv + ((size_t)b * S1) * 384 + 128 + hh * DH);
    v4u g0;
    g0[0] = 1u;                                   // count=1 (valid descriptor)
    g0[1] = (unsigned)(uintptr_t)(&kbuf[0]);      // lds_addr
    g0[2] = (unsigned)(ga & 0xFFFFFFFFull);       // global_addr[31:0]
    g0[3] = (unsigned)((ga >> 32) & 0x01FFFFFFull) | (2u << 30);  // addr[56:32] | type=2
    v8i g1;
    g1[0] = 0x10000;               // data_size=2B (bits 17:16 = 1), no multicast
    g1[1] = (int)(32u << 16);      // tensor_dim0 low16 at bits 63:48
    g1[2] = (int)(129u << 16);     // tensor_dim0 hi=0; tensor_dim1 low16 at bits 95:80
    g1[3] = (int)(32u << 16);      // tensor_dim1 hi=0; tile_dim0=32 at bits 127:112
    g1[4] = 160;                   // tile_dim1=160 (bits 143:128), tile_dim2=0
    g1[5] = 384;                   // tensor_dim0_stride low32
    g1[6] = 0;                     // stride hi, tensor_dim1_stride low16
    g1[7] = 0;
    v4i gz = {0, 0, 0, 0};
#if __clang_major__ >= 23
    v8i g4 = {0, 0, 0, 0, 0, 0, 0, 0};
    __builtin_amdgcn_tensor_load_to_lds(g0, g1, gz, gz, g4, 0);
#else
    __builtin_amdgcn_tensor_load_to_lds(g0, g1, gz, gz, 0);
#endif
    wait_tensor0();
  }
#endif
  __syncthreads();
  v16h aq = frag_ldA_guard(qkv + ((size_t)b * S1 + mt0) * 384 + hh * DH, 384, rowsValid);
  const float scale = 0.17677669529663687f;  // 1/sqrt(32)
  int n = lane & 15, mo = (lane >> 4) << 3;
  for (int nt = w; nt < KP / 16; nt += 8) {
    v8f c = {};
    c = wmma16(aq, frag_ldB(kbuf + (size_t)(nt * 16) * DH, DH), c);
    int j = nt * 16 + n;
    bool valid = (j < S1) && (j == 0 || items[(size_t)b * SN + (j - 1)] != 0);
#pragma unroll
    for (int r = 0; r < 8; ++r)
      sbuf[(mo + r) * KP + nt * 16 + n] = valid ? c[r] * scale : -1e9f;
  }
  __syncthreads();
#pragma unroll
  for (int rr = 0; rr < 2; ++rr) {   // softmax: wave w handles rows 2w, 2w+1
    int m = w * 2 + rr;
    float* srow = sbuf + m * KP;
    float e5[5]; float vmax = -3e38f;
#pragma unroll
    for (int t = 0; t < 5; ++t) { e5[t] = srow[lane + 32 * t]; vmax = fmaxf(vmax, e5[t]); }
    vmax = wred_max(vmax);
    float ssum = 0.f;
#pragma unroll
    for (int t = 0; t < 5; ++t) { e5[t] = __expf(e5[t] - vmax); ssum += e5[t]; }
    float inv = 1.0f / wred_sum(ssum);
    _Float16* prow = pbuf + m * KP;
#pragma unroll
    for (int t = 0; t < 5; ++t) prow[lane + 32 * t] = (_Float16)(e5[t] * inv);
  }
  __syncthreads();
  if (w < 2) {                        // context: N=32 -> 2 n-tiles, K=160 -> 5 steps
    v16h ap[5], bv[5];
#pragma unroll
    for (int ks = 0; ks < KP / 32; ++ks) {
      ap[ks] = frag_ldA(pbuf + ks * 32, KP);
      bv[ks] = frag_ldB(vT + (size_t)(w * 16) * KP + ks * 32, KP);
    }
    v8f c = {};
#pragma unroll
    for (int ks = 0; ks < KP / 32; ++ks) c = wmma16(ap[ks], bv[ks], c);
    _Float16* ob = ctx + ((size_t)b * S1 + mt0) * DM + hh * DH + w * 16 + n;
    if (rowsValid == 16) {
#pragma unroll
      for (int r = 0; r < 8; ++r) ob[(size_t)(mo + r) * DM] = (_Float16)c[r];
    } else {
#pragma unroll
      for (int r = 0; r < 8; ++r) { int m = mo + r; if (m < rowsValid) ob[(size_t)m * DM] = (_Float16)c[r]; }
    }
  }
}

// tok += ctx @ Wo ; h = LN(tok)   (fused)
__global__ void k_proj(const _Float16* __restrict__ ctx, const _Float16* __restrict__ WoT,
                       float* __restrict__ tok, _Float16* __restrict__ hg) {
  __shared__ __align__(16) float tb[16 * DM];
  int b = blockIdx.y, mt0 = blockIdx.x * 16;
  int rowsValid = imin(S1 - mt0, 16);
  int w = threadIdx.x >> 5, lane = threadIdx.x & 31;
  const _Float16* abase = ctx + ((size_t)b * S1 + mt0) * DM;
  v16h afr[4], bfr[4];
#pragma unroll
  for (int ks = 0; ks < 4; ++ks) {
    afr[ks] = frag_ldA_guard(abase + ks * 32, DM, rowsValid);
    bfr[ks] = frag_ldB(WoT + (size_t)(w * 16) * DM + ks * 32, DM);
  }
  v8f c = {};
#pragma unroll
  for (int ks = 0; ks < 4; ++ks) c = wmma16(afr[ks], bfr[ks], c);
  int n = lane & 15, mo = (lane >> 4) << 3;
  const float* resb = tok + ((size_t)b * S1 + mt0) * DM + w * 16 + n;
  if (rowsValid == 16) {
#pragma unroll
    for (int r = 0; r < 8; ++r) {
      int m = mo + r;
      tb[m * DM + w * 16 + n] = c[r] + resb[(size_t)m * DM];
    }
  } else {
#pragma unroll
    for (int r = 0; r < 8; ++r) {
      int m = mo + r;
      float res = (m < rowsValid) ? resb[(size_t)m * DM] : 0.f;
      tb[m * DM + w * 16 + n] = c[r] + res;
    }
  }
  __syncthreads();
#pragma unroll
  for (int rr = 0; rr < 2; ++rr) {
    int m = w * 2 + rr;
    if (mt0 + m >= S1) continue;
    float v4[4]; float sm = 0.f;
#pragma unroll
    for (int t = 0; t < 4; ++t) { v4[t] = tb[m * DM + lane + 32 * t]; sm += v4[t]; }
    float mean = wred_sum(sm) * (1.0f / 128.0f);
    float vs = 0.f;
#pragma unroll
    for (int t = 0; t < 4; ++t) { float d = v4[t] - mean; vs += d * d; }
    float inv = rsqrtf(wred_sum(vs) * (1.0f / 128.0f) + 1e-5f);
#pragma unroll
    for (int t = 0; t < 4; ++t) {
      size_t gi = ((size_t)b * S1 + mt0 + m) * DM + lane + 32 * t;
      tok[gi] = v4[t];
      hg[gi] = (_Float16)((v4[t] - mean) * inv);
    }
  }
}

// tok += gelu(h @ W1) @ W2 ; pass1 keeps 4 independent accumulator chains per wave
__global__ void k_ffn(const _Float16* __restrict__ hg, const _Float16* __restrict__ W1T,
                      const _Float16* __restrict__ W2T, float* __restrict__ tok) {
  __shared__ __align__(16) _Float16 midbuf[16 * DFF];
  int b = blockIdx.y, mt0 = blockIdx.x * 16;
  int rowsValid = imin(S1 - mt0, 16);
  int w = threadIdx.x >> 5, lane = threadIdx.x & 31;
  int n = lane & 15, mo = (lane >> 4) << 3;
  const _Float16* abase = hg + ((size_t)b * S1 + mt0) * DM;
  v16h afr[4];
#pragma unroll
  for (int ks = 0; ks < 4; ++ks) afr[ks] = frag_ldA_guard(abase + ks * 32, DM, rowsValid);
  v8f cc[4] = {};
#pragma unroll
  for (int ks = 0; ks < 4; ++ks) {
    v16h b0 = frag_ldB(W1T + (size_t)((w)      * 16) * DM + ks * 32, DM);
    v16h b1 = frag_ldB(W1T + (size_t)((w + 8)  * 16) * DM + ks * 32, DM);
    v16h b2 = frag_ldB(W1T + (size_t)((w + 16) * 16) * DM + ks * 32, DM);
    v16h b3 = frag_ldB(W1T + (size_t)((w + 24) * 16) * DM + ks * 32, DM);
    cc[0] = wmma16(afr[ks], b0, cc[0]);
    cc[1] = wmma16(afr[ks], b1, cc[1]);
    cc[2] = wmma16(afr[ks], b2, cc[2]);
    cc[3] = wmma16(afr[ks], b3, cc[3]);
  }
#pragma unroll
  for (int q = 0; q < 4; ++q) {
    int nt = w + q * 8;
#pragma unroll
    for (int r = 0; r < 8; ++r) {
      float x = cc[q][r];
      float g = 0.5f * x * (1.0f + tanhf(0.7978845608f * (x + 0.044715f * x * x * x)));
      midbuf[(mo + r) * DFF + nt * 16 + n] = (_Float16)g;
    }
  }
  __syncthreads();
  v8f c2 = {};
#pragma unroll
  for (int kg = 0; kg < 4; ++kg) {    // 16 k-steps, in groups of 4
    v16h a4[4], b4[4];
#pragma unroll
    for (int q = 0; q < 4; ++q) {
      int ks = kg * 4 + q;
      a4[q] = frag_ldA(midbuf + ks * 32, DFF);
      b4[q] = frag_ldB(W2T + (size_t)(w * 16) * DFF + ks * 32, DFF);
    }
#pragma unroll
    for (int q = 0; q < 4; ++q) c2 = wmma16(a4[q], b4[q], c2);
  }
  float* ob = tok + ((size_t)b * S1 + mt0) * DM + w * 16 + n;
  if (rowsValid == 16) {
#pragma unroll
    for (int r = 0; r < 8; ++r) { size_t gi = (size_t)(mo + r) * DM; ob[gi] = ob[gi] + c2[r]; }
  } else {
#pragma unroll
    for (int r = 0; r < 8; ++r) {
      int m = mo + r;
      if (m < rowsValid) { size_t gi = (size_t)m * DM; ob[gi] = ob[gi] + c2[r]; }
    }
  }
}

// pairs -> normalized & weighted factors Af/Bf (f16, row 127 zero pad), den[b]
__global__ void k_pairs(const float* __restrict__ tok, const int* __restrict__ items,
                        const float* __restrict__ ts, _Float16* __restrict__ Af,
                        _Float16* __restrict__ Bf, float* __restrict__ den) {
  __shared__ int cnt;
  int b = blockIdx.x;
  int w = threadIdx.x >> 5, lane = threadIdx.x & 31;
  if (threadIdx.x == 0) cnt = 0;
  __syncthreads();
  for (int l = w; l < LPP; l += 8) {
    if (l >= SN - 1) {   // pad row
#pragma unroll
      for (int t = 0; t < 4; ++t) {
        Af[((size_t)b * LPP + l) * DM + lane + 32 * t] = (_Float16)0.f;
        Bf[((size_t)b * LPP + l) * DM + lane + 32 * t] = (_Float16)0.f;
      }
      continue;
    }
    const float* e0 = tok + ((size_t)b * S1 + 1 + l) * DM;
    const float* e1 = e0 + DM;
    float p4[4]; float s2 = 0.f;
#pragma unroll
    for (int t = 0; t < 4; ++t) { p4[t] = 0.5f * (e0[lane + 32 * t] + e1[lane + 32 * t]); s2 += p4[t] * p4[t]; }
    float nrm = fmaxf(sqrtf(wred_sum(s2)), 1e-8f);
    int pm = (items[(size_t)b * SN + l] != 0 && items[(size_t)b * SN + l + 1] != 0) ? 1 : 0;
    float gap = fmaxf(ts[(size_t)b * SN + l + 1] - ts[(size_t)b * SN + l], 0.f);
    float wgt = fmaxf(__expf(-gap), 1e-3f) * (float)pm;
#pragma unroll
    for (int t = 0; t < 4; ++t) {
      float an = p4[t] / nrm;
      Af[((size_t)b * LPP + l) * DM + lane + 32 * t] = (_Float16)(an * wgt);
      Bf[((size_t)b * LPP + l) * DM + lane + 32 * t] = (_Float16)(an * (float)pm);
    }
    if (lane == 0 && pm) atomicAdd(&cnt, 1);
  }
  __syncthreads();
  if (threadIdx.x == 0) den[b] = fmaxf((float)cnt, 1.0f);
}

// sims: 512x512 GEMM, K=16384. 64x64 block tile, 512 threads = 16 waves, one
// 16x16 tile per wave. Double-buffered async global->LDS staging (ASYNCcnt).
__global__ void __launch_bounds__(512) k_sims(const _Float16* __restrict__ Af,
                                              const _Float16* __restrict__ Bf,
                                              const float* __restrict__ den,
                                              float* __restrict__ sims) {
  __shared__ __align__(16) _Float16 Abuf[2][64 * 32];
  __shared__ __align__(16) _Float16 Bbuf[2][64 * 32];
  int i0 = blockIdx.y * 64, j0 = blockIdx.x * 64;
  int tid = threadIdx.x, w = tid >> 5, lane = tid & 31;
  int mt = w >> 2, nt = w & 3;
  int sh = tid & 255, isB = tid >> 8;          // threads 0-255 stage A, 256-511 stage B
  int row = sh >> 2, c4 = (sh & 3) << 3;
  const _Float16* gsrc =
      (isB ? (Bf + (size_t)(j0 + row) * KSIMS) : (Af + (size_t)(i0 + row) * KSIMS)) + c4;
  _Float16* l0 = isB ? &Bbuf[0][row * 32 + c4] : &Abuf[0][row * 32 + c4];
  _Float16* l1 = isB ? &Bbuf[1][row * 32 + c4] : &Abuf[1][row * 32 + c4];
  v8f acc = {};
  const int KSTEPS = (int)(KSIMS / 32);
  async_ld_b128(l0, gsrc);                     // prologue: stage panel 0
  wait_async0();
  __syncthreads();
  for (int kk = 0; kk < KSTEPS; ++kk) {
    if (kk + 1 < KSTEPS)                       // issue next panel while computing current
      async_ld_b128(((kk + 1) & 1) ? l1 : l0, gsrc + (size_t)(kk + 1) * 32);
    const _Float16* Ab = Abuf[kk & 1];
    const _Float16* Bb = Bbuf[kk & 1];
    v16h a = frag_ldA(Ab + (mt * 16) * 32, 32);
    v16h b = frag_ldB(Bb + (nt * 16) * 32, 32);
    acc = wmma16(a, b, acc);
    wait_async0();
    __syncthreads();
  }
  int n = lane & 15, mo = (lane >> 4) << 3;
#pragma unroll
  for (int r = 0; r < 8; ++r) {
    int ii = i0 + mt * 16 + mo + r;
    sims[(size_t)ii * BN + j0 + nt * 16 + n] = acc[r] / den[ii];
  }
}

// per-row top-5 / bottom-5 + InfoNCE row loss
__global__ void k_topk(const float* __restrict__ sims, const float* __restrict__ tok,
                       float* __restrict__ outv, float* __restrict__ outi,
                       float* __restrict__ rloss) {
  __shared__ float srow[BN];
  __shared__ float srow2[BN];
  __shared__ float bv[256];
  __shared__ int   bi[256];
  __shared__ float topv[5];
  __shared__ int   topi[5];
  __shared__ int   negi[5];
  int i = blockIdx.x, tid = threadIdx.x;
  for (int j = tid; j < BN; j += 256) { float v = sims[(size_t)i * BN + j]; srow[j] = v; srow2[j] = v; }
  __syncthreads();
  for (int t = 0; t < 5; ++t) {   // argmax with lowest-index tie break
    float best = -3e38f; int bidx = BN;
    for (int j = tid; j < BN; j += 256) {
      float v = srow[j];
      if (v > best || (v == best && j < bidx)) { best = v; bidx = j; }
    }
    bv[tid] = best; bi[tid] = bidx;
    __syncthreads();
    for (int off = 128; off; off >>= 1) {
      if (tid < off) {
        float ov = bv[tid + off]; int oi = bi[tid + off];
        if (ov > bv[tid] || (ov == bv[tid] && oi < bi[tid])) { bv[tid] = ov; bi[tid] = oi; }
      }
      __syncthreads();
    }
    if (tid == 0) { topv[t] = bv[0]; topi[t] = bi[0]; srow[bi[0]] = -3e38f; }
    __syncthreads();
  }
  for (int t = 0; t < 5; ++t) {   // argmin with lowest-index tie break
    float best = 3e38f; int bidx = BN;
    for (int j = tid; j < BN; j += 256) {
      float v = srow2[j];
      if (v < best || (v == best && j < bidx)) { best = v; bidx = j; }
    }
    bv[tid] = best; bi[tid] = bidx;
    __syncthreads();
    for (int off = 128; off; off >>= 1) {
      if (tid < off) {
        float ov = bv[tid + off]; int oi = bi[tid + off];
        if (ov < bv[tid] || (ov == bv[tid] && oi < bi[tid])) { bv[tid] = ov; bi[tid] = oi; }
      }
      __syncthreads();
    }
    if (tid == 0) { negi[t] = bi[0]; srow2[bi[0]] = 3e38f; }
    __syncthreads();
  }
  if (tid < 32) {                 // wave 0: loss math on cls vectors
    int lane = tid;
    const float* cls = tok + (size_t)i * S1 * DM;
    float a4[4]; float s2 = 0.f;
#pragma unroll
    for (int t = 0; t < 4; ++t) { a4[t] = cls[lane + 32 * t]; s2 += a4[t] * a4[t]; }
    float na = fmaxf(sqrtf(wred_sum(s2)), 1e-12f);
#pragma unroll
    for (int t = 0; t < 4; ++t) a4[t] /= na;
    float lp[5], lnv[5];
    for (int k = 0; k < 5; ++k) {
      const float* pv = tok + (size_t)topi[k] * S1 * DM;
      float ss = 0.f, dd = 0.f;
#pragma unroll
      for (int t = 0; t < 4; ++t) { float x = pv[lane + 32 * t]; ss += x * x; dd += x * a4[t]; }
      float np = fmaxf(sqrtf(wred_sum(ss)), 1e-12f);
      lp[k] = wred_sum(dd) / np / TAUF;
    }
    for (int k = 0; k < 5; ++k) {
      const float* pv = tok + (size_t)negi[k] * S1 * DM;
      float ss = 0.f, dd = 0.f;
#pragma unroll
      for (int t = 0; t < 4; ++t) { float x = pv[lane + 32 * t]; ss += x * x; dd += x * a4[t]; }
      float np = fmaxf(sqrtf(wred_sum(ss)), 1e-12f);
      lnv[k] = wred_sum(dd) / np / TAUF;
    }
    if (lane == 0) {
      float mp = lp[0], mn = lnv[0];
      for (int k = 1; k < 5; ++k) { mp = fmaxf(mp, lp[k]); mn = fmaxf(mn, lnv[k]); }
      float sp = 0.f, sn = 0.f;
      for (int k = 0; k < 5; ++k) { sp += __expf(lp[k] - mp); sn += __expf(lnv[k] - mn); }
      float lsep = mp + logf(sp), lsen = mn + logf(sn);
      float mD = fmaxf(lsep, lsen);
      float log_denom = mD + logf(__expf(lsep - mD) + __expf(lsen - mD));
      float wk[5]; float wsum = 0.f;
      for (int k = 0; k < 5; ++k) { wk[k] = fmaxf((topv[k] + 1.f) * 0.5f, 1e-12f); wsum += wk[k]; }
      wsum += 1e-12f;
      float xk[5]; float mx = -3e38f;
      for (int k = 0; k < 5; ++k) { xk[k] = logf(wk[k] / wsum + 1e-12f) + lp[k]; mx = fmaxf(mx, xk[k]); }
      float sx = 0.f;
      for (int k = 0; k < 5; ++k) sx += __expf(xk[k] - mx);
      float log_num = mx + logf(sx);
      rloss[i] = -(log_num - log_denom);
      for (int k = 0; k < 5; ++k) {
        outv[(size_t)i * 5 + k] = topv[k];
        outi[(size_t)i * 5 + k] = (float)topi[k];
      }
    }
  }
}

__global__ void k_loss(const float* __restrict__ rloss, float* __restrict__ out0) {
  __shared__ float ss[256];
  int tid = threadIdx.x;
  float v = 0.f;
  for (int j = tid; j < BN; j += 256) v += rloss[j];
  ss[tid] = v;
  __syncthreads();
  for (int off = 128; off; off >>= 1) { if (tid < off) ss[tid] += ss[tid + off]; __syncthreads(); }
  if (tid == 0) out0[0] = ss[0] / (float)BN;
}

// ---------------- launch ----------------
extern "C" void kernel_launch(void* const* d_in, const int* in_sizes, int n_in,
                              void* d_out, int out_size, void* d_ws, size_t ws_size,
                              hipStream_t stream) {
  (void)in_sizes; (void)n_in; (void)out_size; (void)ws_size;
  const int*   items = (const int*)d_in[0];
  const float* ts    = (const float*)d_in[1];
  const float* emb   = (const float*)d_in[2];
  const float* clst  = (const float*)d_in[3];
  const float* pos   = (const float*)d_in[4];
  const float* Wqkv  = (const float*)d_in[5];
  const float* Wo    = (const float*)d_in[6];
  const float* W1    = (const float*)d_in[7];
  const float* W2    = (const float*)d_in[8];
  float* out = (float*)d_out;
  char*  ws  = (char*)d_ws;

  float*    tok   = (float*)(ws + OFF_TOK);
  _Float16* hg    = (_Float16*)(ws + OFF_H);
  _Float16* qkv   = (_Float16*)(ws + OFF_QKV);
  _Float16* ctx   = (_Float16*)(ws + OFF_CTX);
  _Float16* WqkvT = (_Float16*)(ws + OFF_WQ);
  _Float16* WoT   = (_Float16*)(ws + OFF_WO);
  _Float16* W1T   = (_Float16*)(ws + OFF_W1T);
  _Float16* W2T   = (_Float16*)(ws + OFF_W2T);
  _Float16* Af    = (_Float16*)(ws + OFF_AF);
  _Float16* Bf    = (_Float16*)(ws + OFF_BF);
  float*    den   = (float*)(ws + OFF_DEN);
  float*    rloss = (float*)(ws + OFF_RL);

  float* sims = out + 1;
  float* outv = out + 1 + (size_t)BN * BN;
  float* outi = outv + (size_t)BN * 5;

  k_prep <<<768, 256, 0, stream>>>(Wqkv, Wo, W1, W2, WqkvT, WoT, W1T, W2T);
  k_embed<<<BN, 256, 0, stream>>>(items, emb, clst, pos, tok);
  k_ln1  <<<dim3(17, BN), 256, 0, stream>>>(tok, hg);
  k_qkv  <<<dim3(9, BN), 256, 0, stream>>>(hg, WqkvT, qkv);
  k_attn <<<dim3(9, NH, BN), 256, 0, stream>>>(qkv, items, ctx);
  k_proj <<<dim3(9, BN), 256, 0, stream>>>(ctx, WoT, tok, hg);
  k_ffn  <<<dim3(9, BN), 256, 0, stream>>>(hg, W1T, W2T, tok);
  k_pairs<<<BN, 256, 0, stream>>>(tok, items, ts, Af, Bf, den);
  k_sims <<<dim3(8, 8), 512, 0, stream>>>(Af, Bf, den, sims);
  k_topk <<<BN, 256, 0, stream>>>(sims, tok, outv, outi, rloss);
  k_loss <<<1, 256, 0, stream>>>(rloss, out);
}
